// BaseModel_29875792511833
// MI455X (gfx1250) — compile-verified, matrix operand order checked by reading, not run on hardware
//
#include <hip/hip_runtime.h>
#include <hip/hip_bf16.h>

typedef __attribute__((ext_vector_type(16))) _Float16 v16h;
typedef __attribute__((ext_vector_type(8)))  float    v8f;

#define TWO_PI 6.28318530717958647692f

// ---------------------------------------------------------------------------
// Async global->LDS copy (gfx1250 GLOBAL_LOAD_ASYNC_TO_LDS_B32, ASYNCcnt)
// ---------------------------------------------------------------------------
__device__ inline void async_copy_f32(float* dst_lds, const float* src) {
  unsigned loff = (unsigned)(size_t)(__attribute__((address_space(3))) float*)dst_lds;
  asm volatile("global_load_async_to_lds_b32 %0, %1, off"
               :: "v"(loff), "v"(src) : "memory");
}
__device__ inline void async_wait0() {
  asm volatile("s_wait_asynccnt 0" ::: "memory");
}

// ---------------------------------------------------------------------------
// Generic WMMA GEMM:  C[M,N] = act(A[M,K] @ B[N,K]^T + bias) + add
// B stored row-major (N,K) i.e. "weight" layout (out = in @ W.T).
// Each wave owns a 16x64 strip (4 N-tiles): 1 A-tile load feeds 4 WMMAs.
// z-batching: z0 = z % zdiv, z1 = z / zdiv with independent strides.
// act: 0 none, 1 gelu(exact), 2 elu+1, 3 log(x+1e-9)
// ---------------------------------------------------------------------------
struct GemmP {
  const float* A; const float* B; float* C;
  const float* bias; const float* add;
  int M, N, K, lda, ldb, ldc, act, zdiv;
  long As0, As1, Bs0, Bs1, Cs0, Cs1, Ds0, Ds1;
};

__device__ inline v16h ld_tile_f32_as_f16(const float* p, int ld, int lane) {
  // 16x32 f16 operand tile layout (wave32):
  // lane r (0..15): row=r, K in {0..7,16..23}; lane r+16: row=r, K in {8..15,24..31}
  int r = lane & 15, hh = lane >> 4;
  const float* q = p + (long)r * ld + hh * 8;
  v16h out;
#pragma unroll
  for (int i = 0; i < 8; ++i) out[i] = (_Float16)q[i];
#pragma unroll
  for (int i = 0; i < 8; ++i) out[8 + i] = (_Float16)q[16 + i];
  return out;
}

__global__ void __launch_bounds__(128) gemm_wmma_f16(GemmP p) {
  int wave = threadIdx.x >> 5;
  int lane = threadIdx.x & 31;
  int ntiles = p.N >> 4;
  int nt0 = (blockIdx.x * 4 + wave) * 4;      // first of up to 4 N-tiles
  int mt  = blockIdx.y;
  if (nt0 >= ntiles) return;
  int z  = blockIdx.z;
  int z0 = z % p.zdiv, z1 = z / p.zdiv;
  const float* A = p.A + (long)z0 * p.As0 + (long)z1 * p.As1 + (long)mt * 16 * p.lda;
  const float* B = p.B + (long)z0 * p.Bs0 + (long)z1 * p.Bs1;

  v8f acc[4] = {{}, {}, {}, {}};
#pragma unroll 2
  for (int k = 0; k < p.K; k += 32) {
    v16h a = ld_tile_f32_as_f16(A + k, p.lda, lane);
#pragma unroll
    for (int tn = 0; tn < 4; ++tn) {
      int nt = nt0 + tn;
      if (nt < ntiles) {   // wave-uniform branch: EXEC stays all-ones for WMMA
        v16h b = ld_tile_f32_as_f16(B + (long)nt * 16 * p.ldb + k, p.ldb, lane);
        acc[tn] = __builtin_amdgcn_wmma_f32_16x16x32_f16(false, a, false, b,
                                                         (short)0, acc[tn], false, false);
      }
    }
  }

  // C/D layout: lane L: n = L&15, m = (L>>4)*8 + v
  int n  = lane & 15;
  int mb = (lane >> 4) * 8;
  long cbase = (long)z0 * p.Cs0 + (long)z1 * p.Cs1;
  long dbase = (long)z0 * p.Ds0 + (long)z1 * p.Ds1;
#pragma unroll
  for (int tn = 0; tn < 4; ++tn) {
    int nt = nt0 + tn;
    if (nt >= ntiles) break;
    int col = nt * 16 + n;
    float bv = p.bias ? p.bias[col] : 0.0f;
#pragma unroll
    for (int v = 0; v < 8; ++v) {
      int row = mt * 16 + mb + v;
      float val = acc[tn][v] + bv;
      if (p.act == 1)      val = 0.5f * val * (1.0f + erff(val * 0.70710678118f));
      else if (p.act == 2) val = (val > 0.0f) ? (val + 1.0f) : expf(val);   // elu(x)+1
      else if (p.act == 3) val = logf(val + 1e-9f);
      if (p.add) val += p.add[dbase + (long)row * p.ldc + col];
      p.C[cbase + (long)row * p.ldc + col] = val;
    }
  }
}

// ---------------------------------------------------------------------------
// Mel filterbank (transposed: fbT[128][1056], cols >=1025 zero-padded)
// ---------------------------------------------------------------------------
__global__ void melfb_kernel(float* fbT) {
  int i = blockIdx.x * 256 + threadIdx.x;
  if (i >= 128 * 1056) return;
  int m = i / 1056, f = i % 1056;
  float val = 0.0f;
  if (f <= 1024) {
    float freq   = (float)f * (22050.0f / 1024.0f);
    float melmax = 2595.0f * log10f(1.0f + 22050.0f / 700.0f);
    float step   = melmax / 129.0f;
    const float LN10 = 2.30258509299404568402f;
    float p0 = 700.0f * (expf((float)(m)     * step / 2595.0f * LN10) - 1.0f);
    float p1 = 700.0f * (expf((float)(m + 1) * step / 2595.0f * LN10) - 1.0f);
    float p2 = 700.0f * (expf((float)(m + 2) * step / 2595.0f * LN10) - 1.0f);
    float down = (freq - p0) / (p1 - p0);
    float up   = (p2 - freq) / (p2 - p1);
    val = fmaxf(0.0f, fminf(down, up));
  }
  fbT[i] = val;
}

// ---------------------------------------------------------------------------
// 2048-pt FFT per frame (one block), Hann window, reflect pad, power spectrum
// spec: 512 frames x 1056 (cols 1025..1055 zeroed)
// ---------------------------------------------------------------------------
__global__ void __launch_bounds__(256) melspec_fft(const float* x, const float* y,
                                                   float* spec, int sb) {
  __shared__ float re[2048];
  __shared__ float im[2048];
  const int N = 262144;
  int frame = blockIdx.x;
  int tid = threadIdx.x;
  const float* wav = ((sb < 2) ? x : y) + (long)(sb & 1) * N;

  for (int j = tid; j < 2048; j += 256) {
    long t  = (long)frame * 512 + j - 1024;
    long ti = (t < 0) ? -t : ((t >= N) ? (2L * N - 2 - t) : t);
    float w = 0.5f * (1.0f - __cosf(TWO_PI * (float)j / 2048.0f));
    unsigned br = __brev((unsigned)j) >> (32 - 11);
    re[br] = wav[ti] * w;
    im[br] = 0.0f;
  }
  __syncthreads();

  for (int s = 1; s <= 11; ++s) {
    int m = 1 << s, mh = m >> 1;
    for (int idx = tid; idx < 1024; idx += 256) {
      int grp = idx / mh, pos = idx - grp * mh;
      int i0 = grp * m + pos, i1 = i0 + mh;
      float ang = -TWO_PI * (float)pos / (float)m;
      float sn, cs;
      __sincosf(ang, &sn, &cs);
      float tr = re[i1] * cs - im[i1] * sn;
      float ti = re[i1] * sn + im[i1] * cs;
      re[i1] = re[i0] - tr;  im[i1] = im[i0] - ti;
      re[i0] = re[i0] + tr;  im[i0] = im[i0] + ti;
    }
    __syncthreads();
  }

  for (int b = tid; b < 1056; b += 256) {
    float v = (b <= 1024) ? (re[b] * re[b] + im[b] * im[b]) : 0.0f;
    spec[(long)frame * 1056 + b] = v;
  }
}

// ---------------------------------------------------------------------------
// Transpose P (H,32,128) -> PT (H,128,32)  (weight format for WMMA GEMM)
// ---------------------------------------------------------------------------
__global__ void transpose_P(const float* P, float* PT) {
  int i = blockIdx.x * 256 + threadIdx.x;
  if (i >= 8 * 32 * 128) return;
  int h = i >> 12, rem = i & 4095;
  int d = rem >> 7, f = rem & 127;
  PT[(h * 128 + f) * 32 + d] = P[i];
}

// ---------------------------------------------------------------------------
// Causal linear attention scan, one block per (b,h). 128x32 state in LDS.
// Per-step operand rows staged into LDS with GLOBAL_LOAD_ASYNC_TO_LDS_B32.
// normalize=1: write attn/(denom+1e-8) into (b,n,256) at col h*32
// normalize=0: write raw attn into (bh,32,512) + denom (bh,512)
// ---------------------------------------------------------------------------
__global__ void __launch_bounds__(256) lin_attn(const float* qp, const float* kp,
                                                const float* v, int ldv, long vzs, int vcol0,
                                                float* out_flat, float* out_t,
                                                float* denom_out, int normalize) {
  __shared__ float S[128 * 32];
  __shared__ float kc[128];
  __shared__ float qs[128], ks[128], vs[32];
  __shared__ float red[256], redd[128];
  int bh = blockIdx.x, b = bh >> 3, h = bh & 7;
  int t = threadIdx.x;
  const float* qpb = qp + (long)bh * 512 * 128;
  const float* kpb = kp + (long)bh * 512 * 128;
  const float* vb  = v + (long)b * vzs + vcol0 + h * 32;

  for (int i = t; i < 128 * 32; i += 256) S[i] = 0.0f;
  if (t < 128) kc[t] = 0.0f;
  __syncthreads();

  for (int n = 0; n < 512; ++n) {
    if (t < 128) {
      async_copy_f32(&qs[t], qpb + n * 128 + t);
      async_copy_f32(&ks[t], kpb + n * 128 + t);
    } else if (t < 160) {
      async_copy_f32(&vs[t - 128], vb + (long)n * ldv + (t - 128));
    }
    async_wait0();
    __syncthreads();

    { // state update: S[f][d] += k[f]*v[d]; kcum
      int f = t >> 1;
      int d0 = (t & 1) * 16;
      float kf = ks[f];
#pragma unroll
      for (int j = 0; j < 16; ++j) S[f * 32 + d0 + j] += kf * vs[d0 + j];
      if ((t & 1) == 0) kc[f] += kf;
    }
    __syncthreads();

    { // attn partials + denom partials
      int d = t & 31, fg = t >> 5;
      float partial = 0.0f;
#pragma unroll
      for (int j = 0; j < 16; ++j) {
        int f = fg * 16 + j;
        partial += qs[f] * S[f * 32 + d];
      }
      red[t] = partial;
      if (t < 128) redd[t] = qs[t] * kc[t];
    }
    __syncthreads();
    if (t < 32) redd[t] = redd[t] + redd[t + 32] + redd[t + 64] + redd[t + 96];
    __syncthreads();
    if (t == 0) {
      float s = 0.0f;
      for (int i = 0; i < 32; ++i) s += redd[i];
      redd[0] = s;
    }
    __syncthreads();
    if (t < 32) {
      float a = 0.0f;
#pragma unroll
      for (int g = 0; g < 8; ++g) a += red[g * 32 + t];
      float den = redd[0];
      if (normalize) {
        out_flat[((long)(b * 512 + n)) * 256 + h * 32 + t] = a / (den + 1e-8f);
      } else {
        out_t[((long)bh * 32 + t) * 512 + n] = a;
        if (t == 0) denom_out[(long)bh * 512 + n] = den;
      }
    }
    __syncthreads();
  }
}

// ---------------------------------------------------------------------------
// conv1d SAME over seq, channels 32, layout (16, 32, 512)
// ---------------------------------------------------------------------------
__global__ void conv1d_k(const float* in, const float* w, const float* bias,
                         float* out, int ksz) {
  int i = blockIdx.x * 256 + threadIdx.x;
  if (i >= 16 * 32 * 512) return;
  int n = i & 511, co = (i >> 9) & 31, bh = i >> 14;
  int pad = (ksz - 1) / 2;
  float s = bias[co];
  for (int ci = 0; ci < 32; ++ci) {
    const float* row = in + ((long)bh * 32 + ci) * 512;
    const float* wr  = w + (co * 32 + ci) * ksz;
    for (int j = 0; j < ksz; ++j) {
      int pos = n + j - pad;
      if (pos >= 0 && pos < 512) s += row[pos] * wr[j];
    }
  }
  out[i] = s;
}

__global__ void norm_transpose(const float* zt, const float* denom, float* out) {
  int i = blockIdx.x * 256 + threadIdx.x;
  if (i >= 16 * 32 * 512) return;
  int n = i & 511, d = (i >> 9) & 31, bh = i >> 14;
  int b = bh >> 3, h = bh & 7;
  out[((long)(b * 512 + n)) * 256 + h * 32 + d] = zt[i] / (denom[(long)bh * 512 + n] + 1e-8f);
}

// ---------------------------------------------------------------------------
// LayerNorm over last dim (256), one block per row
// ---------------------------------------------------------------------------
__global__ void __launch_bounds__(256) layernorm(const float* x, const float* g,
                                                 const float* bta, float* y) {
  __shared__ float red[256];
  int row = blockIdx.x, t = threadIdx.x;
  float v = x[(long)row * 256 + t];
  red[t] = v; __syncthreads();
  for (int s = 128; s > 0; s >>= 1) { if (t < s) red[t] += red[t + s]; __syncthreads(); }
  float mu = red[0] * (1.0f / 256.0f);
  __syncthreads();
  float dv = v - mu;
  red[t] = dv * dv; __syncthreads();
  for (int s = 128; s > 0; s >>= 1) { if (t < s) red[t] += red[t + s]; __syncthreads(); }
  float var = red[0] * (1.0f / 256.0f);
  y[(long)row * 256 + t] = dv * rsqrtf(var + 1e-5f) * g[t] + bta[t];
}

__global__ void seq_mean(const float* x, float* o) {
  int i = blockIdx.x * 256 + threadIdx.x;
  if (i >= 512) return;
  int b = i >> 8, c = i & 255;
  float s = 0.0f;
  for (int n = 0; n < 512; ++n) s += x[((long)(b * 512 + n)) * 256 + c];
  o[i] = s * (1.0f / 512.0f);
}

// ---------------------------------------------------------------------------
// Final head: LN -> FFN(256->1024->256, gelu) -> LN -> linear(1) -> sigmoid
// ---------------------------------------------------------------------------
__global__ void __launch_bounds__(256) final_head(const float* o,
    const float* ln1g, const float* ln1b,
    const float* W1, const float* b1, const float* W2, const float* b2,
    const float* ln2g, const float* ln2b,
    const float* Wo, const float* bo, float* out) {
  __shared__ float hbuf[2][256];
  __shared__ float h1[2][1024];
  __shared__ float h2[2][256];
  __shared__ float red[256];
  int t = threadIdx.x;

  for (int b = 0; b < 2; ++b) {
    float v = o[b * 256 + t];
    red[t] = v; __syncthreads();
    for (int s = 128; s > 0; s >>= 1) { if (t < s) red[t] += red[t + s]; __syncthreads(); }
    float mu = red[0] * (1.0f / 256.0f); __syncthreads();
    float dv = v - mu;
    red[t] = dv * dv; __syncthreads();
    for (int s = 128; s > 0; s >>= 1) { if (t < s) red[t] += red[t + s]; __syncthreads(); }
    float var = red[0] * (1.0f / 256.0f); __syncthreads();
    hbuf[b][t] = dv * rsqrtf(var + 1e-5f) * ln1g[t] + ln1b[t];
  }
  __syncthreads();

  for (int b = 0; b < 2; ++b)
    for (int j = t; j < 1024; j += 256) {
      float s = b1[j];
      for (int c = 0; c < 256; ++c) s += hbuf[b][c] * W1[j * 256 + c];
      h1[b][j] = 0.5f * s * (1.0f + erff(s * 0.70710678118f));
    }
  __syncthreads();

  for (int b = 0; b < 2; ++b) {
    float s = b2[t];
    for (int j = 0; j < 1024; ++j) s += h1[b][j] * W2[t * 1024 + j];
    h2[b][t] = s;
  }
  __syncthreads();

  for (int b = 0; b < 2; ++b) {
    float v = h2[b][t];
    red[t] = v; __syncthreads();
    for (int s = 128; s > 0; s >>= 1) { if (t < s) red[t] += red[t + s]; __syncthreads(); }
    float mu = red[0] * (1.0f / 256.0f); __syncthreads();
    float dv = v - mu;
    red[t] = dv * dv; __syncthreads();
    for (int s = 128; s > 0; s >>= 1) { if (t < s) red[t] += red[t + s]; __syncthreads(); }
    float var = red[0] * (1.0f / 256.0f); __syncthreads();
    float ln = dv * rsqrtf(var + 1e-5f) * ln2g[t] + ln2b[t];
    red[t] = ln * Wo[t]; __syncthreads();
    for (int s = 128; s > 0; s >>= 1) { if (t < s) red[t] += red[t + s]; __syncthreads(); }
    if (t == 0) out[b] = 1.0f / (1.0f + expf(-(red[0] + bo[0])));
    __syncthreads();
  }
}

// ---------------------------------------------------------------------------
// Host orchestration
// ---------------------------------------------------------------------------
static void gemm(hipStream_t s, const float* A, const float* B, float* C,
                 const float* bias, const float* add,
                 int M, int N, int K, int lda, int ldb, int ldc, int act,
                 int nz = 1, int zdiv = 1,
                 long As0 = 0, long As1 = 0, long Bs0 = 0, long Bs1 = 0,
                 long Cs0 = 0, long Cs1 = 0, long Ds0 = 0, long Ds1 = 0) {
  GemmP p;
  p.A = A; p.B = B; p.C = C; p.bias = bias; p.add = add;
  p.M = M; p.N = N; p.K = K; p.lda = lda; p.ldb = ldb; p.ldc = ldc;
  p.act = act; p.zdiv = zdiv;
  p.As0 = As0; p.As1 = As1; p.Bs0 = Bs0; p.Bs1 = Bs1;
  p.Cs0 = Cs0; p.Cs1 = Cs1; p.Ds0 = Ds0; p.Ds1 = Ds1;
  dim3 g((unsigned)((N / 16 + 15) / 16), (unsigned)(M / 16), (unsigned)nz);
  gemm_wmma_f16<<<g, 128, 0, s>>>(p);
}

extern "C" void kernel_launch(void* const* d_in, const int* in_sizes, int n_in,
                              void* d_out, int out_size, void* d_ws, size_t ws_size,
                              hipStream_t stream) {
  auto IN = [&](int i) { return (const float*)d_in[i]; };
  const float* xin = IN(0);
  const float* yin = IN(1);
  const float* projW = IN(2);
  const float* projb = IN(3);
  const float* posemb = IN(4);

  float* ws = (float*)d_ws;
  size_t off = 0;
  auto alloc = [&](size_t n) { float* p = ws + off; off += n; return p; };
  float* melfbT = alloc(128 * 1056);
  float* spec   = alloc(512 * 1056);
  float* mels   = alloc(4 * 512 * 128);
  float* xe     = alloc(2 * 512 * 256);
  float* ye     = alloc(2 * 512 * 256);
  float* qkv    = alloc(2 * 512 * 768);
  float* qp     = alloc(2 * 8 * 512 * 128);
  float* kp     = alloc(2 * 8 * 512 * 128);
  float* attnf  = alloc(1024 * 256);
  float* tmp1   = alloc(1024 * 256);
  float* tmp2   = alloc(1024 * 256);
  float* h1buf  = alloc(1024 * 1024);
  float* PT     = alloc(8 * 128 * 32);
  float* qbuf   = alloc(1024 * 256);
  float* kbuf   = alloc(1024 * 256);
  float* vbuf   = alloc(1024 * 256);
  float* at0    = alloc(16 * 32 * 512);
  float* at1    = alloc(16 * 32 * 512);
  float* den    = alloc(16 * 512);
  float* obuf   = alloc(512);

  // ---- mel spectrograms + log-mel (spec @ melfbT via WMMA, log epilogue) ----
  melfb_kernel<<<(128 * 1056 + 255) / 256, 256, 0, stream>>>(melfbT);
  for (int sb = 0; sb < 4; ++sb) {
    melspec_fft<<<512, 256, 0, stream>>>(xin, yin, spec, sb);
    gemm(stream, spec, melfbT, mels + (size_t)sb * 512 * 128, nullptr, nullptr,
         512, 128, 1056, 1056, 1056, 128, /*log*/3);
  }

  // ---- audio embed: mels @ projW.T + projb + pos_emb[:512] ----
  gemm(stream, mels, projW, xe, projb, posemb, 512, 256, 128, 128, 128, 256, 0,
       2, 2, /*As0 b*/512 * 128, 0, 0, 0, /*Cs0 b*/512 * 256, 0, /*Ds0*/0, 0);
  gemm(stream, mels + 2 * 512 * 128, projW, ye, projb, posemb, 512, 256, 128, 128, 128, 256, 0,
       2, 2, 512 * 128, 0, 0, 0, 512 * 256, 0, 0, 0);

  // ---- performer stacks (stream 0 -> xe/block1, stream 1 -> ye/block2) ----
  for (int s = 0; s < 2; ++s) {
    int base = 5 + s * 10;
    const float* P    = IN(base + 0);
    const float* Wqkv = IN(base + 1);
    const float* Wout = IN(base + 2);
    const float* bout = IN(base + 3);
    const float* lng  = IN(base + 4);
    const float* lnb  = IN(base + 5);
    const float* W1   = IN(base + 6);
    const float* b1   = IN(base + 7);
    const float* W2   = IN(base + 8);
    const float* b2   = IN(base + 9);
    float* cur = (s == 0) ? xe : ye;

    for (int l = 0; l < 8; ++l) {
      transpose_P<<<(8 * 32 * 128 + 255) / 256, 256, 0, stream>>>(P + (size_t)l * 8 * 32 * 128, PT);
      // qkv = cur @ Wqkv.T   (1024 x 768 x 256)
      gemm(stream, cur, Wqkv + (size_t)l * 768 * 256, qkv, nullptr, nullptr,
           1024, 768, 256, 256, 256, 768, 0);
      // qp = elu(q_h @ P_h)+1 per (b,h): z0=h (stride 32), z1=b (stride 512*768)
      gemm(stream, qkv + 0, PT, qp, nullptr, nullptr, 512, 128, 32, 768, 32, 128, /*elu+1*/2,
           16, 8, 32, 512L * 768, 128 * 32, 0, 512L * 128, 8L * 512 * 128, 0, 0);
      gemm(stream, qkv + 256, PT, kp, nullptr, nullptr, 512, 128, 32, 768, 32, 128, 2,
           16, 8, 32, 512L * 768, 128 * 32, 0, 512L * 128, 8L * 512 * 128, 0, 0);
      // causal linear attention (v lives in qkv at col 512+h*32)
      lin_attn<<<16, 256, 0, stream>>>(qp, kp, qkv, 768, 512L * 768, 512,
                                       attnf, nullptr, nullptr, 1);
      // out proj + residual
      gemm(stream, attnf, Wout + (size_t)l * 256 * 256, tmp1,
           bout + (size_t)l * 256, cur, 1024, 256, 256, 256, 256, 256, 0);
      layernorm<<<1024, 256, 0, stream>>>(tmp1, lng + (size_t)l * 256, lnb + (size_t)l * 256, tmp2);
      // FFN
      gemm(stream, tmp2, W1 + (size_t)l * 1024 * 256, h1buf,
           b1 + (size_t)l * 1024, nullptr, 1024, 1024, 256, 256, 256, 1024, /*gelu*/1);
      gemm(stream, h1buf, W2 + (size_t)l * 256 * 1024, tmp1,
           b2 + (size_t)l * 256, tmp2, 1024, 256, 1024, 1024, 1024, 256, 0);
      layernorm<<<1024, 256, 0, stream>>>(tmp1, lng + (size_t)l * 256, lnb + (size_t)l * 256, cur);
    }
  }

  // ---- music cross block ----
  const float* Pm    = IN(25);
  const float* Wq    = IN(26);
  const float* Wk    = IN(27);
  const float* Wv    = IN(28);
  const float* Woutm = IN(29);
  const float* boutm = IN(30);
  const float* lngm  = IN(31);
  const float* lnbm  = IN(32);
  const float* W1m   = IN(33);
  const float* b1m   = IN(34);
  const float* W2m   = IN(35);
  const float* b2m   = IN(36);

  transpose_P<<<(8 * 32 * 128 + 255) / 256, 256, 0, stream>>>(Pm, PT);
  gemm(stream, xe, Wq, qbuf, nullptr, nullptr, 1024, 256, 256, 256, 256, 256, 0);
  gemm(stream, ye, Wk, kbuf, nullptr, nullptr, 1024, 256, 256, 256, 256, 256, 0);
  gemm(stream, ye, Wv, vbuf, nullptr, nullptr, 1024, 256, 256, 256, 256, 256, 0);
  gemm(stream, qbuf, PT, qp, nullptr, nullptr, 512, 128, 32, 256, 32, 128, 2,
       16, 8, 32, 512L * 256, 128 * 32, 0, 512L * 128, 8L * 512 * 128, 0, 0);
  gemm(stream, kbuf, PT, kp, nullptr, nullptr, 512, 128, 32, 256, 32, 128, 2,
       16, 8, 32, 512L * 256, 128 * 32, 0, 512L * 128, 8L * 512 * 128, 0, 0);
  lin_attn<<<16, 256, 0, stream>>>(qp, kp, vbuf, 256, 512L * 256, 0,
                                   nullptr, at0, den, 0);
  // conv chain 3,5,7,9 (ping-pong)
  conv1d_k<<<(16 * 32 * 512 + 255) / 256, 256, 0, stream>>>(at0, IN(37), IN(38), at1, 3);
  conv1d_k<<<(16 * 32 * 512 + 255) / 256, 256, 0, stream>>>(at1, IN(39), IN(40), at0, 5);
  conv1d_k<<<(16 * 32 * 512 + 255) / 256, 256, 0, stream>>>(at0, IN(41), IN(42), at1, 7);
  conv1d_k<<<(16 * 32 * 512 + 255) / 256, 256, 0, stream>>>(at1, IN(43), IN(44), at0, 9);
  norm_transpose<<<(16 * 32 * 512 + 255) / 256, 256, 0, stream>>>(at0, den, attnf);

  gemm(stream, attnf, Woutm, tmp1, boutm, xe, 1024, 256, 256, 256, 256, 256, 0);
  layernorm<<<1024, 256, 0, stream>>>(tmp1, lngm, lnbm, tmp2);
  gemm(stream, tmp2, W1m, h1buf, b1m, nullptr, 1024, 1024, 256, 256, 256, 1024, 1);
  gemm(stream, h1buf, W2m, tmp1, b2m, tmp2, 1024, 256, 1024, 1024, 1024, 256, 0);
  layernorm<<<1024, 256, 0, stream>>>(tmp1, lngm, lnbm, tmp2);

  // ---- head ----
  seq_mean<<<2, 256, 0, stream>>>(tmp2, obuf);
  final_head<<<1, 256, 0, stream>>>(obuf, IN(45), IN(46), IN(47), IN(48),
                                    IN(49), IN(50), IN(51), IN(52),
                                    IN(53), IN(54), (float*)d_out);
}
